// Sender_85547158602076
// MI455X (gfx1250) — matrix operand order repaired and checked by reading, not compiled, not run on hardware
//
#include <hip/hip_runtime.h>
#include <hip/hip_bf16.h>

// ---------------------------------------------------------------------------
// Problem constants (match reference)
// ---------------------------------------------------------------------------
#define BB   512
#define INN  64
#define EE   512
#define HH   1024
#define VV   1024
#define LLEN 32
#define LL1  33        // MAX_LEN + 1 (zero-padded last column)
#define G4H  4096      // 4*H

typedef __attribute__((ext_vector_type(16))) _Float16 v16h;
typedef __attribute__((ext_vector_type(8)))  float    v8f;
typedef __attribute__((ext_vector_type(4)))  unsigned int v4u;
typedef __attribute__((ext_vector_type(4)))  unsigned int u32x4;
typedef __attribute__((ext_vector_type(8)))  int      i32x8;
typedef __attribute__((ext_vector_type(4)))  int      i32x4;

union V16H { v16h v; v4u q[2]; };

#if defined(__HIP_DEVICE_COMPILE__) && __has_builtin(__builtin_amdgcn_tensor_load_to_lds)
#define HAVE_TDM 1
#else
#define HAVE_TDM 0
#endif

// ---------------------------------------------------------------------------
// TDM issue: async-DMA a 64-row x 32-col f16 tile (rows stride K elements)
// from global into LDS.  D# built per CDNA5 ISA 8.3-8.6:
//   group0: count=1 | lds_addr | global_addr[56:0] | type=2
//   group1: data_size=2B, tensor_dim0=K, tensor_dim1=64, tile=32x64,
//           tensor_dim0_stride=K
//   groups 2/3 (+ trailing group): zero (2-D tensor)
// 6-arg builtin on this toolchain:
//   (uint32x4 g0, int32x8 g1, int32x4 g2, int32x4 g3, int32x8 g4, i32 cpol)
// ---------------------------------------------------------------------------
__device__ __forceinline__ void tdm_issue_b_tile(const _Float16* gsrc, int K,
                                                 const _Float16* ldst)
{
#if HAVE_TDM
    unsigned long long ga = (unsigned long long)(uintptr_t)gsrc;
    unsigned lds = (unsigned)(uintptr_t)ldst;   // low 32 bits = LDS byte address
    u32x4 g0 = { 1u,                                   // count=1
                 lds,
                 (unsigned)(ga & 0xFFFFFFFFu),
                 (unsigned)((ga >> 32) & 0x01FFFFFFu) | 0x80000000u }; // type=2
    unsigned d0 = (unsigned)K;         // tensor_dim0 (row length, elements)
    unsigned d1 = 64u;                 // tensor_dim1 (rows)
    unsigned long long s0 = (unsigned long long)(unsigned)K; // dim0 stride
    i32x8 g1 = { (int)(1u << 16),                                  // data_size=2B
                 (int)((d0 & 0xFFFFu) << 16),                      // dim0[15:0]
                 (int)(((d0 >> 16) & 0xFFFFu) | ((d1 & 0xFFFFu) << 16)),
                 (int)(((d1 >> 16) & 0xFFFFu) | (32u << 16)),      // tile_dim0=32
                 (int)(64u),                                       // tile_dim1=64
                 (int)(s0 & 0xFFFFFFFFu),
                 (int)((s0 >> 32) & 0xFFFFu),
                 0 };
    i32x4 gz4 = { 0, 0, 0, 0 };
    i32x8 gz8 = { 0, 0, 0, 0, 0, 0, 0, 0 };
    __builtin_amdgcn_tensor_load_to_lds(g0, g1, gz4, gz4, gz8, 0);
#else
    (void)gsrc; (void)K; (void)ldst;
#endif
}

#define WAVES_PER_BLOCK 8
#define GEMM_BLOCK 256

// ---------------------------------------------------------------------------
// Fused dual-operand GEMM:  C[M,N] = A1@W1^T (+ A2@W2^T) + bias
// Block tile: 128 rows (8 waves x 16) x 64 cols.  Weight (B) 64x32 K-chunks
// are TDM-DMA'd into double-buffered LDS and shared by all 8 waves; the DMA
// for chunk i+1 overlaps WMMA on chunk i (s_wait_tensorcnt + barrier).
// A fragments: direct per-lane global_load_b128 (L2-resident activations),
// packed in contiguous k order (k = 16*half + s), same as B.  B fragments: all four 16-col tiles
// are ds_load_b128'd up front so the four WMMAs overlap the LDS reads.
// Grid: x = N/64, y = M/128.
// ---------------------------------------------------------------------------
__global__ void __launch_bounds__(GEMM_BLOCK)
gemm_wmma_f16(const _Float16* __restrict__ A1, const _Float16* __restrict__ W1, int K1,
              const _Float16* __restrict__ A2, const _Float16* __restrict__ W2, int K2,
              const float* __restrict__ bias, float* __restrict__ C, int N)
{
    __shared__ _Float16 smem[2][64 * 32];

    const int lane = threadIdx.x & 31;
    const int wave = threadIdx.x >> 5;
    const int half = lane >> 4;      // 0: lanes 0-15, 1: lanes 16-31
    const int lidx = lane & 15;

    const int n0 = blockIdx.x << 6;              // 64 output columns per block
    const int m0 = (blockIdx.y << 7) + (wave << 4);  // 16 rows per wave

    const int c1 = K1 >> 5;
    const int c2 = (A2 != nullptr) ? (K2 >> 5) : 0;
    const int nc = c1 + c2;

    v8f acc[4] = {};

#if HAVE_TDM
    if (wave == 0) {
        tdm_issue_b_tile(W1 + (size_t)n0 * K1, K1, &smem[0][0]);
    }
#endif

    for (int ci = 0; ci < nc; ++ci) {
        // operand select (uniform)
        const _Float16* Ab; const _Float16* Wb; int Kc, kk;
        if (ci < c1) { Ab = A1; Wb = W1; Kc = K1; kk = ci << 5; }
        else         { Ab = A2; Wb = W2; Kc = K2; kk = (ci - c1) << 5; }

#if HAVE_TDM
        const int buf = ci & 1;
        if (wave == 0) {
            __builtin_amdgcn_s_wait_tensorcnt(0);   // chunk ci landed in LDS
        }
        __syncthreads();                            // publish; prev compute done
        if (wave == 0 && (ci + 1) < nc) {           // prefetch chunk ci+1
            const _Float16* Wn; int Kn, kn;
            if (ci + 1 < c1) { Wn = W1; Kn = K1; kn = (ci + 1) << 5; }
            else             { Wn = W2; Kn = K2; kn = (ci + 1 - c1) << 5; }
            tdm_issue_b_tile(Wn + (size_t)n0 * Kn + kn, Kn, &smem[buf ^ 1][0]);
        }
#else
        const int buf = 0;
        __syncthreads();                            // prev compute done
        {   // cooperative 64x32 f16 tile load: 256 threads x 8 f16
            const int row = threadIdx.x >> 2;
            const int seg = (threadIdx.x & 3) << 3;
            *(v4u*)&smem[0][(row << 5) + seg] =
                *(const v4u*)(Wb + (size_t)(n0 + row) * Kc + kk + seg);
        }
        __syncthreads();
#endif
        // ---- A fragment (16x32, k = 16*half + s) -------------------------------
        const _Float16* __restrict__ ar =
            Ab + (size_t)(m0 + lidx) * Kc + kk + half * 16;
        V16H a;
        a.q[0] = *(const v4u*)(ar);
        a.q[1] = *(const v4u*)(ar + 8);
        __builtin_prefetch(ar + 32, 0, 1);

        // ---- load all four B fragments first (8 ds_load_b128 as a clause) -
        const _Float16* bp = &smem[buf][(lidx << 5) + (half << 4)];
        V16H b0, b1, b2, b3;
        b0.q[0] = *(const v4u*)(bp);         b0.q[1] = *(const v4u*)(bp + 8);
        b1.q[0] = *(const v4u*)(bp + 512);   b1.q[1] = *(const v4u*)(bp + 520);
        b2.q[0] = *(const v4u*)(bp + 1024);  b2.q[1] = *(const v4u*)(bp + 1032);
        b3.q[0] = *(const v4u*)(bp + 1536);  b3.q[1] = *(const v4u*)(bp + 1544);

        // ---- 4 WMMA column tiles ------------------------------------------
        acc[0] = __builtin_amdgcn_wmma_f32_16x16x32_f16(
            false, a.v, false, b0.v, (short)0, acc[0], false, false);
        acc[1] = __builtin_amdgcn_wmma_f32_16x16x32_f16(
            false, a.v, false, b1.v, (short)0, acc[1], false, false);
        acc[2] = __builtin_amdgcn_wmma_f32_16x16x32_f16(
            false, a.v, false, b2.v, (short)0, acc[2], false, false);
        acc[3] = __builtin_amdgcn_wmma_f32_16x16x32_f16(
            false, a.v, false, b3.v, (short)0, acc[3], false, false);
    }

    // ---- bias + store (C layout: col = lane&15, row = r + 8*half) ---------
#pragma unroll
    for (int nt = 0; nt < 4; ++nt) {
        const int col = n0 + (nt << 4) + lidx;
        const float bv = bias ? bias[col] : 0.0f;
        float* __restrict__ cp = C + (size_t)(m0 + half * 8) * N + col;
#pragma unroll
        for (int r = 0; r < 8; ++r)
            cp[(size_t)r * N] = acc[nt][r] + bv;
    }
}

// ---------------------------------------------------------------------------
// Elementwise LSTM cell update: gates[B,4H] (i,f,g,o blocks) + c -> c', h'
// ---------------------------------------------------------------------------
__global__ void lstm_elem(const float* __restrict__ gates,
                          float* __restrict__ c, float* __restrict__ h,
                          _Float16* __restrict__ h16, int n /* B*H */)
{
    int idx = blockIdx.x * blockDim.x + threadIdx.x;
    if (idx >= n) return;
    int b = idx >> 10;          // / H
    int j = idx & (HH - 1);     // % H
    const float* gr = gates + (size_t)b * G4H;
    float gi = gr[j];
    float gf = gr[HH + j];
    float gg = gr[2 * HH + j];
    float go = gr[3 * HH + j];
    gi = 1.0f / (1.0f + __expf(-gi));
    gf = 1.0f / (1.0f + __expf(-gf));
    go = 1.0f / (1.0f + __expf(-go));
    gg = tanhf(gg);
    float cc = gf * c[idx] + gi * gg;
    c[idx] = cc;
    float hh = go * tanhf(cc);
    h[idx] = hh;
    h16[idx] = (_Float16)hh;
}

// ---------------------------------------------------------------------------
// LayerNorm over H, output f16 (feeds the logits GEMM). One block per row.
// ---------------------------------------------------------------------------
__global__ void layernorm_f16(const float* __restrict__ h,
                              const float* __restrict__ g, const float* __restrict__ b,
                              _Float16* __restrict__ out)
{
    __shared__ float s1[256], s2[256];
    const int row = blockIdx.x;
    const int tid = threadIdx.x;
    const float* x = h + (size_t)row * HH;
    float a = 0.f, q = 0.f;
    for (int j = tid; j < HH; j += 256) { float v = x[j]; a += v; q += v * v; }
    s1[tid] = a; s2[tid] = q;
    __syncthreads();
    for (int o = 128; o > 0; o >>= 1) {
        if (tid < o) { s1[tid] += s1[tid + o]; s2[tid] += s2[tid + o]; }
        __syncthreads();
    }
    const float mu = s1[0] * (1.0f / HH);
    const float var = s2[0] * (1.0f / HH) - mu * mu;
    const float rs = rsqrtf(var + 1e-5f);
    for (int j = tid; j < HH; j += 256)
        out[(size_t)row * HH + j] = (_Float16)((x[j] - mu) * rs * g[j] + b[j]);
}

// ---------------------------------------------------------------------------
// Per-row log-softmax stats: argmax, logprob(argmax) = -logZ, entropy.
// ---------------------------------------------------------------------------
__global__ void softmax_rows(const float* __restrict__ S, int t,
                             int* __restrict__ seq_out, float* __restrict__ lp_out,
                             float* __restrict__ ent_out, int* __restrict__ sym_buf)
{
    __shared__ float rf[256], rf2[256];
    __shared__ int   ri[256];
    const int b = blockIdx.x;
    const int tid = threadIdx.x;
    const float* s = S + (size_t)b * VV;

    float m = -3.402823466e38f; int mi = 0;
    for (int j = tid; j < VV; j += 256) {
        float v = s[j];
        if (v > m) { m = v; mi = j; }
    }
    rf[tid] = m; ri[tid] = mi;
    __syncthreads();
    for (int o = 128; o > 0; o >>= 1) {
        if (tid < o) {
            if (rf[tid + o] > rf[tid] ||
                (rf[tid + o] == rf[tid] && ri[tid + o] < ri[tid])) {
                rf[tid] = rf[tid + o]; ri[tid] = ri[tid + o];
            }
        }
        __syncthreads();
    }
    const float mx = rf[0];
    const int amax = ri[0];
    __syncthreads();

    float z = 0.f, zs = 0.f;
    for (int j = tid; j < VV; j += 256) {
        float e = __expf(s[j] - mx);
        z += e; zs += e * s[j];
    }
    rf[tid] = z; rf2[tid] = zs;
    __syncthreads();
    for (int o = 128; o > 0; o >>= 1) {
        if (tid < o) { rf[tid] += rf[tid + o]; rf2[tid] += rf2[tid + o]; }
        __syncthreads();
    }
    if (tid == 0) {
        const float Z = rf[0];
        const float logZ = __logf(Z);
        sym_buf[b] = amax;
        seq_out[(size_t)b * LL1 + t] = amax;
        lp_out [(size_t)b * LL1 + t] = -logZ;                  // lp at argmax
        ent_out[(size_t)b * LL1 + t] = mx + logZ - rf2[0] / Z; // -sum p*lp
    }
}

// next input = emb[sym]  (f16 embedding table)
__global__ void gather_emb(const _Float16* __restrict__ emb16,
                           const int* __restrict__ sym, _Float16* __restrict__ cur)
{
    const int b = blockIdx.x;
    const _Float16* src = emb16 + (size_t)sym[b] * EE;
    for (int j = threadIdx.x; j < EE; j += blockDim.x)
        cur[(size_t)b * EE + j] = src[j];
}

// ---------------------------------------------------------------------------
// Prep kernels
// ---------------------------------------------------------------------------
__global__ void cvt_f32_f16(const float* __restrict__ src, _Float16* __restrict__ dst, int n)
{
    for (int i = blockIdx.x * blockDim.x + threadIdx.x; i < n; i += gridDim.x * blockDim.x)
        dst[i] = (_Float16)src[i];
}
__global__ void add_bias2(const float* __restrict__ a, const float* __restrict__ b,
                          float* __restrict__ dst, int n)
{
    int i = blockIdx.x * blockDim.x + threadIdx.x;
    if (i < n) dst[i] = a[i] + b[i];
}
__global__ void zero_f32(float* __restrict__ p, int n)
{
    for (int i = blockIdx.x * blockDim.x + threadIdx.x; i < n; i += gridDim.x * blockDim.x)
        p[i] = 0.0f;
}
__global__ void zero_f16(_Float16* __restrict__ p, int n)
{
    for (int i = blockIdx.x * blockDim.x + threadIdx.x; i < n; i += gridDim.x * blockDim.x)
        p[i] = (_Float16)0.0f;
}

// ---------------------------------------------------------------------------
// Host orchestration
// ---------------------------------------------------------------------------
extern "C" void kernel_launch(void* const* d_in, const int* in_sizes, int n_in,
                              void* d_out, int out_size, void* d_ws, size_t ws_size,
                              hipStream_t stream)
{
    const float* x      = (const float*)d_in[0];
    const float* w_in   = (const float*)d_in[1];
    const float* b_in   = (const float*)d_in[2];
    const float* w_ih0  = (const float*)d_in[3];
    const float* w_hh0  = (const float*)d_in[4];
    const float* b_ih0  = (const float*)d_in[5];
    const float* b_hh0  = (const float*)d_in[6];
    const float* w_ih1  = (const float*)d_in[7];
    const float* w_hh1  = (const float*)d_in[8];
    const float* b_ih1  = (const float*)d_in[9];
    const float* b_hh1  = (const float*)d_in[10];
    const float* ln_g   = (const float*)d_in[11];
    const float* ln_b   = (const float*)d_in[12];
    const float* w_out  = (const float*)d_in[13];
    const float* b_out  = (const float*)d_in[14];
    const float* emb    = (const float*)d_in[15];

    // ---- workspace carve-out (256B aligned slots) -------------------------
    char* wp = (char*)d_ws;
    size_t off = 0;
    auto take = [&](size_t bytes) -> char* {
        char* p = wp + off;
        off += (bytes + 255) & ~(size_t)255;
        return p;
    };
    _Float16* x16     = (_Float16*)take((size_t)BB * INN * 2);
    _Float16* w_in16  = (_Float16*)take((size_t)EE * INN * 2);
    _Float16* w_ih016 = (_Float16*)take((size_t)G4H * EE * 2);
    _Float16* w_hh016 = (_Float16*)take((size_t)G4H * HH * 2);
    _Float16* w_ih116 = (_Float16*)take((size_t)G4H * HH * 2);
    _Float16* w_hh116 = (_Float16*)take((size_t)G4H * HH * 2);
    _Float16* w_out16 = (_Float16*)take((size_t)VV * HH * 2);
    _Float16* emb16   = (_Float16*)take((size_t)VV * EE * 2);
    float*    bias0   = (float*)take((size_t)G4H * 4);
    float*    bias1   = (float*)take((size_t)G4H * 4);
    float*    inp_f   = (float*)take((size_t)BB * EE * 4);
    _Float16* cur16   = (_Float16*)take((size_t)BB * EE * 2);
    float*    h0      = (float*)take((size_t)BB * HH * 4);
    float*    c0      = (float*)take((size_t)BB * HH * 4);
    float*    h1      = (float*)take((size_t)BB * HH * 4);
    float*    c1      = (float*)take((size_t)BB * HH * 4);
    _Float16* h0_16   = (_Float16*)take((size_t)BB * HH * 2);
    _Float16* h1_16   = (_Float16*)take((size_t)BB * HH * 2);
    _Float16* hn16    = (_Float16*)take((size_t)BB * HH * 2);
    float*    gates   = (float*)take((size_t)BB * G4H * 4);
    float*    logits  = (float*)take((size_t)BB * VV * 4);
    int*      symb    = (int*)take((size_t)BB * 4);
    (void)ws_size; (void)n_in; (void)in_sizes; (void)out_size;

    int*   seq_out = (int*)d_out;                       // [B, 33] int32 bits
    float* lp_out  = (float*)d_out + (size_t)BB * LL1;  // [B, 33]
    float* ent_out = lp_out + (size_t)BB * LL1;         // [B, 33]

    const int CT = 256;
    auto cvt = [&](const float* s, _Float16* d, int n) {
        int g = (n + CT - 1) / CT; if (g > 2048) g = 2048;
        hipLaunchKernelGGL(cvt_f32_f16, dim3(g), dim3(CT), 0, stream, s, d, n);
    };

    // ---- one-time (per call) prep ----------------------------------------
    cvt(x,     x16,     BB * INN);
    cvt(w_in,  w_in16,  EE * INN);
    cvt(w_ih0, w_ih016, G4H * EE);
    cvt(w_hh0, w_hh016, G4H * HH);
    cvt(w_ih1, w_ih116, G4H * HH);
    cvt(w_hh1, w_hh116, G4H * HH);
    cvt(w_out, w_out16, VV * HH);
    cvt(emb,   emb16,   VV * EE);
    hipLaunchKernelGGL(add_bias2, dim3((G4H + CT - 1) / CT), dim3(CT), 0, stream,
                       b_ih0, b_hh0, bias0, G4H);
    hipLaunchKernelGGL(add_bias2, dim3((G4H + CT - 1) / CT), dim3(CT), 0, stream,
                       b_ih1, b_hh1, bias1, G4H);
    hipLaunchKernelGGL(zero_f32, dim3(512), dim3(CT), 0, stream, h0, BB * HH);
    hipLaunchKernelGGL(zero_f32, dim3(512), dim3(CT), 0, stream, c0, BB * HH);
    hipLaunchKernelGGL(zero_f32, dim3(512), dim3(CT), 0, stream, h1, BB * HH);
    hipLaunchKernelGGL(zero_f32, dim3(512), dim3(CT), 0, stream, c1, BB * HH);
    hipLaunchKernelGGL(zero_f16, dim3(512), dim3(CT), 0, stream, h0_16, BB * HH);
    hipLaunchKernelGGL(zero_f16, dim3(512), dim3(CT), 0, stream, h1_16, BB * HH);
    hipLaunchKernelGGL(zero_f32, dim3(512), dim3(CT), 0, stream,
                       (float*)d_out, 3 * BB * LL1);   // includes padded last column

    // ---- input projection: inp = x @ w_in^T + b_in -> cur16 ---------------
    hipLaunchKernelGGL(gemm_wmma_f16, dim3(EE / 64, BB / 128), dim3(GEMM_BLOCK),
                       0, stream,
                       x16, w_in16, INN,
                       (const _Float16*)nullptr, (const _Float16*)nullptr, 0,
                       b_in, inp_f, EE);
    cvt(inp_f, cur16, BB * EE);

    // ---- 32 serial decode steps ------------------------------------------
    const int ewBlocks = (BB * HH + CT - 1) / CT;   // 2048

    for (int t = 0; t < LLEN; ++t) {
        // cell 0: gates = cur @ w_ih0^T + h0 @ w_hh0^T + bias0
        hipLaunchKernelGGL(gemm_wmma_f16, dim3(G4H / 64, BB / 128), dim3(GEMM_BLOCK),
                           0, stream,
                           cur16, w_ih016, EE, h0_16, w_hh016, HH,
                           bias0, gates, G4H);
        hipLaunchKernelGGL(lstm_elem, dim3(ewBlocks), dim3(CT), 0, stream,
                           gates, c0, h0, h0_16, BB * HH);
        // cell 1: gates = h0 @ w_ih1^T + h1 @ w_hh1^T + bias1
        hipLaunchKernelGGL(gemm_wmma_f16, dim3(G4H / 64, BB / 128), dim3(GEMM_BLOCK),
                           0, stream,
                           h0_16, w_ih116, HH, h1_16, w_hh116, HH,
                           bias1, gates, G4H);
        hipLaunchKernelGGL(lstm_elem, dim3(ewBlocks), dim3(CT), 0, stream,
                           gates, c1, h1, h1_16, BB * HH);
        // layernorm(h1) -> hn16
        hipLaunchKernelGGL(layernorm_f16, dim3(BB), dim3(256), 0, stream,
                           h1, ln_g, ln_b, hn16);
        // logits = hn @ w_out^T + b_out
        hipLaunchKernelGGL(gemm_wmma_f16, dim3(VV / 64, BB / 128), dim3(GEMM_BLOCK),
                           0, stream,
                           hn16, w_out16, HH,
                           (const _Float16*)nullptr, (const _Float16*)nullptr, 0,
                           b_out, logits, VV);
        // softmax stats + argmax + outputs
        hipLaunchKernelGGL(softmax_rows, dim3(BB), dim3(256), 0, stream,
                           logits, t, seq_out, lp_out, ent_out, symb);
        // next input embedding
        hipLaunchKernelGGL(gather_emb, dim3(BB), dim3(128), 0, stream,
                           emb16, symb, cur16);
    }
}
